// ViT_9045201126013
// MI455X (gfx1250) — matrix operand order repaired and checked by reading, not compiled, not run on hardware
//
#include <hip/hip_runtime.h>
#include <hip/hip_bf16.h>

typedef __attribute__((ext_vector_type(16))) _Float16 v16h;
typedef __attribute__((ext_vector_type(8)))  float    v8f;
typedef __attribute__((ext_vector_type(2)))  float    v2f;

#define N_IMG   512
#define C3      3
#define SDIM    224
#define PSZ     14
#define NP      16
#define INPUT_D 588
#define KPAD    608      // 19 * 32
#define NKC     19       // KPAD / 32
#define HID     8
#define SEQ     257
#define SEQP    272      // 17 * 16
#define NTILE   17
#define HEADS   2
#define DH      4
#define MLP_D   32
#define LNUM    2
#define OUT_DIM 10
#define EPS_LN  1e-5f

// ---------------------------------------------------------------------------
// Kernel A: fused patchify + token embedding (the 308 MB streaming pass).
// One WG = one image * one row of 16 patches. Coalesced 224-wide row loads,
// f32->f16 convert, and scatter DIRECTLY into WMMA-fragment-ordered LDS
// buffers, so the compute wave reads each operand as one aligned 32-byte
// vector load (2x ds_load_b128) per v_wmma_f32_16x16x32_f16.
//
// A-frag order (16-bit 16x32, ISA 7.12.2): lane = M + 16*hi where
//   hi = (kr>>3)&1,  e = kr - 8*hi - 8*(kr>>4)        (kr = K mod 32)
// B-frag order (16-bit 32x16): same K order as A: lane = N + 16*hi, e = kr - 8*hi - 8*(kr>>4)
// ---------------------------------------------------------------------------
__global__ __launch_bounds__(256)
void patch_embed_kernel(const float* __restrict__ images,
                        const float* __restrict__ Wm,
                        const float* __restrict__ bm,
                        const float* __restrict__ pe,
                        float* __restrict__ xbuf) {
    __shared__ __attribute__((aligned(32))) _Float16 Afrag[NKC * 32 * 16];
    __shared__ __attribute__((aligned(32))) _Float16 Bfrag[NKC * 32 * 16];

    const int py  = blockIdx.x;
    const int img = blockIdx.y;
    const int tid = threadIdx.x;

    // ---- Wm -> B fragments (forward loop over every cell: no write races,
    //      zero-pads N in [8,16) and K in [588,608) in the same pass) ----
    for (int cidx = tid; cidx < NKC * 32 * 16; cidx += 256) {
        int kc   = cidx >> 9;            // /512
        int rem  = cidx & 511;
        int lane = rem >> 4;
        int e    = rem & 15;
        int n    = lane & 15;
        int k    = kc * 32 + ((lane >> 4) << 3) + e + ((e >> 3) << 3);   // kr = hi*8 + e + 8*(e>>3)
        float v  = (k < INPUT_D && n < HID) ? Wm[k * HID + n] : 0.f;
        Bfrag[cidx] = (_Float16)v;
    }

    // ---- image tile -> A fragments. Consecutive threads read consecutive
    //      columns of a 224-wide image row: fully coalesced, touched once. ----
    for (int e = tid; e < C3 * PSZ * SDIM; e += 256) {
        int c    = e / (PSZ * SDIM);
        int rem  = e - c * (PSZ * SDIM);
        int iy   = rem / SDIM;
        int xc   = rem - iy * SDIM;
        int px   = xc / PSZ;
        int ix   = xc - px * PSZ;
        int k    = c * (PSZ * PSZ) + iy * PSZ + ix;
        float v  = images[(((size_t)img * C3 + c) * SDIM + (py * PSZ + iy)) * SDIM + xc];
        int kc   = k >> 5;
        int kr   = k & 31;
        int hi   = (kr >> 3) & 1;
        int ea   = kr - 8 * hi - 8 * (kr >> 4);
        int lane = px + 16 * hi;
        Afrag[((kc * 32 + lane) << 4) + ea] = (_Float16)v;
    }
    // zero-pad A for k in [588, 608)  (disjoint cells from the loop above)
    for (int i = tid; i < 16 * (KPAD - INPUT_D); i += 256) {
        int px   = i / (KPAD - INPUT_D);
        int k    = INPUT_D + i % (KPAD - INPUT_D);
        int kc   = k >> 5;
        int kr   = k & 31;
        int hi   = (kr >> 3) & 1;
        int ea   = kr - 8 * hi - 8 * (kr >> 4);
        int lane = px + 16 * hi;
        Afrag[((kc * 32 + lane) << 4) + ea] = (_Float16)0.f;
    }

    __syncthreads();

    if (tid < 32) {                         // wave 0: 19 clean WMMAs
        const int lane = tid;
        const int m15  = lane & 15;
        const int hi   = lane >> 4;
        v8f acc = {};
        for (int kc = 0; kc < NKC; ++kc) {
            v16h a = *(const v16h*)&Afrag[(kc * 32 + lane) << 4];
            v16h b = *(const v16h*)&Bfrag[(kc * 32 + lane) << 4];
            acc = __builtin_amdgcn_wmma_f32_16x16x32_f16(false, a, false, b,
                                                         (short)0, acc, false, false);
        }
        // D layout: vgpr r -> M = r + 8*hi, N = lane&15
        if (m15 < HID) {
#pragma unroll
            for (int r = 0; r < 8; ++r) {
                int m   = r + 8 * hi;                 // patch px
                int seq = 1 + py * 16 + m;
                xbuf[((size_t)img * SEQP + seq) * HID + m15] =
                    acc[r] + bm[m15] + pe[seq * HID + m15];
            }
        }
    }
}

// ---------------------------------------------------------------------------
// Kernel B: cls token + positional embedding for seq 0.
// ---------------------------------------------------------------------------
__global__ void cls_pe_kernel(const int* __restrict__ labels,
                              const float* __restrict__ cls_table,
                              const float* __restrict__ pe,
                              float* __restrict__ xbuf) {
    int t = blockIdx.x * blockDim.x + threadIdx.x;
    if (t < N_IMG * HID) {
        int img = t >> 3, n = t & 7;
        xbuf[((size_t)img * SEQP) * HID + n] = cls_table[labels[img] * HID + n] + pe[n];
    }
}

// ---------------------------------------------------------------------------
// Kernel C: one transformer layer, one workgroup per image, fully LDS
// resident. Attention = flash-style online softmax with
// v_wmma_f32_16x16x4_f32 (DH == 4 exactly matches K=4).
// ---------------------------------------------------------------------------
__global__ __launch_bounds__(256)
void vit_layer_kernel(float* __restrict__ xbuf,
                      const float* __restrict__ ln1_g, const float* __restrict__ ln1_b,
                      const float* __restrict__ Wq,   const float* __restrict__ bq,
                      const float* __restrict__ Wk,   const float* __restrict__ bk,
                      const float* __restrict__ Wv,   const float* __restrict__ bv,
                      const float* __restrict__ ln2_g, const float* __restrict__ ln2_b,
                      const float* __restrict__ W1,   const float* __restrict__ b1,
                      const float* __restrict__ W2,   const float* __restrict__ b2,
                      int l) {
    __shared__ float xs[SEQP][HID];
    __shared__ float qh[HEADS][SEQP][DH];
    __shared__ float kh[HEADS][SEQP][DH];
    __shared__ float vh[HEADS][SEQP][DH];
    __shared__ float pbuf[8][16][17];       // per-wave P-tile bounce buffer

    const int img  = blockIdx.x;
    const int tid  = threadIdx.x;
    const int lane = tid & 31;
    const int wave = tid >> 5;
    float* xg = xbuf + (size_t)img * SEQP * HID;

    // ---- load x, LN1, QKV projections (per row; HID=8 is row-local) ----
    for (int r = tid; r < SEQP; r += 256) {
        float xr[HID];
        if (r < SEQ) {
#pragma unroll
            for (int n = 0; n < HID; ++n) xr[n] = xg[r * HID + n];
        } else {
#pragma unroll
            for (int n = 0; n < HID; ++n) xr[n] = 0.f;
        }
#pragma unroll
        for (int n = 0; n < HID; ++n) xs[r][n] = xr[n];

        if (r < SEQ) {
            float mu = 0.f;
#pragma unroll
            for (int n = 0; n < HID; ++n) mu += xr[n];
            mu *= (1.f / HID);
            float var = 0.f;
#pragma unroll
            for (int n = 0; n < HID; ++n) { float d = xr[n] - mu; var += d * d; }
            var *= (1.f / HID);
            float inv = rsqrtf(var + EPS_LN);
            float h[HID];
#pragma unroll
            for (int n = 0; n < HID; ++n)
                h[n] = (xr[n] - mu) * inv * ln1_g[l * HID + n] + ln1_b[l * HID + n];

            for (int hd = 0; hd < HEADS; ++hd) {
                const float* wqp = Wq + ((l * HEADS + hd) * DH) * DH;
                const float* wkp = Wk + ((l * HEADS + hd) * DH) * DH;
                const float* wvp = Wv + ((l * HEADS + hd) * DH) * DH;
#pragma unroll
                for (int e = 0; e < DH; ++e) {
                    float q = bq[(l * HEADS + hd) * DH + e];
                    float k = bk[(l * HEADS + hd) * DH + e];
                    float v = bv[(l * HEADS + hd) * DH + e];
#pragma unroll
                    for (int d = 0; d < DH; ++d) {
                        float hv = h[hd * DH + d];
                        q += hv * wqp[d * DH + e];
                        k += hv * wkp[d * DH + e];
                        v += hv * wvp[d * DH + e];
                    }
                    qh[hd][r][e] = q * 0.5f;    // fold 1/sqrt(DH)
                    kh[hd][r][e] = k;
                    vh[hd][r][e] = v;
                }
            }
        } else {
            for (int hd = 0; hd < HEADS; ++hd)
#pragma unroll
                for (int e = 0; e < DH; ++e) {
                    qh[hd][r][e] = 0.f; kh[hd][r][e] = 0.f; vh[hd][r][e] = 0.f;
                }
        }
    }
    __syncthreads();

    // ---- flash attention: one wave per (head, 16-row query tile) ----
    const int m15 = lane & 15;
    const int hi  = lane >> 4;
    for (int it = wave; it < HEADS * NTILE; it += 8) {
        int head = it / NTILE;
        int i    = it - head * NTILE;

        // A frag (f32 16x4): lane = M, K = v + 2*hi
        v2f qa;
        qa.x = qh[head][i * 16 + m15][2 * hi + 0];
        qa.y = qh[head][i * 16 + m15][2 * hi + 1];

        float mrun[8], lrun[8];
#pragma unroll
        for (int r = 0; r < 8; ++r) { mrun[r] = -1e30f; lrun[r] = 0.f; }
        v8f oacc = {};

        for (int j = 0; j < NTILE; ++j) {
            // B frag (f32 4x16): lane = N (key col), K = v + 2*hi
            int keyN = j * 16 + m15;
            v2f kb2;
            kb2.x = kh[head][keyN][2 * hi + 0];
            kb2.y = kh[head][keyN][2 * hi + 1];

            v8f s = {};
            s = __builtin_amdgcn_wmma_f32_16x16x4_f32(false, qa, false, kb2,
                                                      (short)0, s, false, false);
            bool cvalid = (keyN < SEQ);
            float pr[8];
#pragma unroll
            for (int r = 0; r < 8; ++r) {
                float sv = cvalid ? s[r] : -1e30f;
                float mx = sv;                           // row max across 16-lane group
                mx = fmaxf(mx, __shfl_xor(mx, 1, 32));
                mx = fmaxf(mx, __shfl_xor(mx, 2, 32));
                mx = fmaxf(mx, __shfl_xor(mx, 4, 32));
                mx = fmaxf(mx, __shfl_xor(mx, 8, 32));
                float mnew  = fmaxf(mrun[r], mx);
                float scale = __expf(mrun[r] - mnew);
                float p     = cvalid ? __expf(sv - mnew) : 0.f;
                float psum  = p;                         // row sum
                psum += __shfl_xor(psum, 1, 32);
                psum += __shfl_xor(psum, 2, 32);
                psum += __shfl_xor(psum, 4, 32);
                psum += __shfl_xor(psum, 8, 32);
                lrun[r] = lrun[r] * scale + psum;
                mrun[r] = mnew;
                oacc[r] *= scale;
                pr[r]   = p;
            }
            // D layout -> LDS: M = r + 8*hi, N = lane&15 (same-wave bounce)
#pragma unroll
            for (int r = 0; r < 8; ++r)
                pbuf[wave][r + 8 * hi][m15] = pr[r];
            asm volatile("s_wait_dscnt 0" ::: "memory");

            // P(16x16) @ V(16x4) as 4 chunks of v_wmma_f32_16x16x4_f32.
            // V loads use clamped in-bounds addresses + value select (cndmask),
            // NOT predicated loads, to keep EXEC all-ones around the WMMAs.
#pragma unroll
            for (int kc = 0; kc < 4; ++kc) {
                v2f pa, vb;
                pa.x = pbuf[wave][m15][kc * 4 + 2 * hi + 0];
                pa.y = pbuf[wave][m15][kc * 4 + 2 * hi + 1];
                int keyr = j * 16 + kc * 4 + 2 * hi;
                float vx = vh[head][keyr + 0][m15 & 3];
                float vy = vh[head][keyr + 1][m15 & 3];
                vb.x = (m15 < DH) ? vx : 0.f;
                vb.y = (m15 < DH) ? vy : 0.f;
                oacc = __builtin_amdgcn_wmma_f32_16x16x4_f32(false, pa, false, vb,
                                                             (short)0, oacc, false, false);
            }
        }
        // residual add: o element (M = r+8*hi, N = lane&15 < DH)
        if (m15 < DH) {
#pragma unroll
            for (int r = 0; r < 8; ++r) {
                int seq = i * 16 + r + 8 * hi;
                if (seq < SEQ) {
                    float invl = 1.f / fmaxf(lrun[r], 1e-20f);
                    xs[seq][head * DH + m15] += oacc[r] * invl;
                }
            }
        }
    }
    __syncthreads();

    // ---- LN2 + MLP (8 -> 32 -> 8, exact GELU) + residual + writeback ----
    for (int r = tid; r < SEQ; r += 256) {
        float xr[HID];
#pragma unroll
        for (int n = 0; n < HID; ++n) xr[n] = xs[r][n];
        float mu = 0.f;
#pragma unroll
        for (int n = 0; n < HID; ++n) mu += xr[n];
        mu *= (1.f / HID);
        float var = 0.f;
#pragma unroll
        for (int n = 0; n < HID; ++n) { float d = xr[n] - mu; var += d * d; }
        var *= (1.f / HID);
        float inv = rsqrtf(var + EPS_LN);
        float h2[HID];
#pragma unroll
        for (int n = 0; n < HID; ++n)
            h2[n] = (xr[n] - mu) * inv * ln2_g[l * HID + n] + ln2_b[l * HID + n];

        float o[HID];
#pragma unroll
        for (int n = 0; n < HID; ++n) o[n] = b2[l * HID + n];
        for (int m = 0; m < MLP_D; ++m) {
            float t = b1[l * MLP_D + m];
#pragma unroll
            for (int n = 0; n < HID; ++n) t += h2[n] * W1[(l * HID + n) * MLP_D + m];
            float g = 0.5f * t * (1.f + erff(t * 0.70710678118f));   // exact GELU
#pragma unroll
            for (int n = 0; n < HID; ++n) o[n] += g * W2[(l * MLP_D + m) * HID + n];
        }
#pragma unroll
        for (int n = 0; n < HID; ++n) xg[r * HID + n] = xr[n] + o[n];
    }
}

// ---------------------------------------------------------------------------
// Kernel D: classifier head on the cls token.
// ---------------------------------------------------------------------------
__global__ void head_kernel(const float* __restrict__ xbuf,
                            const float* __restrict__ Wf,
                            const float* __restrict__ bf,
                            float* __restrict__ out) {
    int t = blockIdx.x * blockDim.x + threadIdx.x;
    if (t < N_IMG * OUT_DIM) {
        int img = t / OUT_DIM, d = t - img * OUT_DIM;
        float acc = bf[d];
#pragma unroll
        for (int n = 0; n < HID; ++n)
            acc += xbuf[(size_t)img * SEQP * HID + n] * Wf[n * OUT_DIM + d];
        out[t] = acc;
    }
}

extern "C" void kernel_launch(void* const* d_in, const int* in_sizes, int n_in,
                              void* d_out, int out_size, void* d_ws, size_t ws_size,
                              hipStream_t stream) {
    const float* images    = (const float*)d_in[0];
    const int*   labels    = (const int*)  d_in[1];
    const float* Wm        = (const float*)d_in[2];
    const float* bm        = (const float*)d_in[3];
    const float* cls_table = (const float*)d_in[4];
    const float* pe        = (const float*)d_in[5];
    const float* ln1_g     = (const float*)d_in[6];
    const float* ln1_b     = (const float*)d_in[7];
    const float* Wq        = (const float*)d_in[8];
    const float* bq        = (const float*)d_in[9];
    const float* Wk        = (const float*)d_in[10];
    const float* bk        = (const float*)d_in[11];
    const float* Wv        = (const float*)d_in[12];
    const float* bv        = (const float*)d_in[13];
    const float* ln2_g     = (const float*)d_in[14];
    const float* ln2_b     = (const float*)d_in[15];
    const float* W1        = (const float*)d_in[16];
    const float* b1        = (const float*)d_in[17];
    const float* W2        = (const float*)d_in[18];
    const float* b2        = (const float*)d_in[19];
    const float* Wf        = (const float*)d_in[20];
    const float* bf        = (const float*)d_in[21];
    float* out  = (float*)d_out;
    float* xbuf = (float*)d_ws;     // [512][272][8] f32 = ~4.5 MB

    patch_embed_kernel<<<dim3(NP, N_IMG), 256, 0, stream>>>(images, Wm, bm, pe, xbuf);
    cls_pe_kernel<<<(N_IMG * HID + 255) / 256, 256, 0, stream>>>(labels, cls_table, pe, xbuf);
    for (int l = 0; l < LNUM; ++l)
        vit_layer_kernel<<<N_IMG, 256, 0, stream>>>(xbuf, ln1_g, ln1_b, Wq, bq, Wk, bk,
                                                    Wv, bv, ln2_g, ln2_b, W1, b1, W2, b2, l);
    head_kernel<<<(N_IMG * OUT_DIM + 255) / 256, 256, 0, stream>>>(xbuf, Wf, bf, out);
}
